// SlotAttention_35098472743560
// MI455X (gfx1250) — matrix operand order repaired and checked by reading, not compiled, not run on hardware
//
#include <hip/hip_runtime.h>

// ---------------------------------------------------------------------------
// Slot Attention fused pipeline for gfx1250 (MI455X), wave32 + WMMA bf16.
// All WMMA fragments are contiguous b128-pair loads; every weight matrix is
// pre-swizzled ONCE into B-fragment order (bf16) by k5_prep. Workspace:
//   kbuf  : bf16[128*4096*64]  (64 MB)  LN(x) @ Wk, token-major
//   vT    : bf16[128*64*4096]  (64 MB)  LN(x) @ Wv, d-major (transposed)
//   qf    : bf16[128*2*512]    (256 KB) per-iter q in B-fragment order
//   Wkf/Wvf/Wqf : bf16[4096]   (8 KB ea)  weight B-fragments
//   gKf/gRf     : bf16[12288]  (24 KB ea) GRU kernels, B-fragments
//   W1f         : bf16[8192]   (16 KB)    MLP W1 B-fragments
//   W2f         : bf16[8192]   (16 KB)    MLP W2 B-fragments
//   upd   : f32 [128*16*64]    (512 KB) raw attn^T @ v partial sums
//   csum  : f32 [128*16]       (8 KB)   per-slot attention column sums
// ---------------------------------------------------------------------------

#define DSLOT 64
#define NS 11
#define SPAD 16
#define HID 128
#define NTOK 4096
#define NBATCH 128
#define NSPLIT 8

typedef __bf16 bf16;
typedef __attribute__((ext_vector_type(16))) bf16  v16bf;
typedef __attribute__((ext_vector_type(8)))  bf16  v8bf;
typedef __attribute__((ext_vector_type(8)))  float v8f;
typedef __attribute__((ext_vector_type(4)))  unsigned int v4u;

struct B32 { v4u lo, hi; };

static __device__ inline v8f wmma_bf16(v16bf a, v16bf b, v8f c) {
  return __builtin_amdgcn_wmma_f32_16x16x32_bf16(false, a, false, b, (short)0, c,
                                                 false, false);
}

// A-fragment (16x32, row-major source): two contiguous 16-byte runs per lane.
static __device__ inline v16bf load_a_frag_c(const bf16* base, int stride, int kbase, int lane) {
  int r = lane & 15, hf = lane >> 4;
  const bf16* p = base + (size_t)r * stride + kbase + hf * 16;
  B32 t;
  t.lo = *(const v4u*)p;
  t.hi = *(const v4u*)(p + 8);
  return __builtin_bit_cast(v16bf, t);
}

// Fragment stored pre-swizzled: 32 contiguous bytes per lane.
static __device__ inline v16bf load_frag32(const bf16* p) {
  B32 t;
  t.lo = *(const v4u*)p;
  t.hi = *(const v4u*)(p + 8);
  return __builtin_bit_cast(v16bf, t);
}

static __device__ inline v4u pack_bf16x8(v8f c) {
  v8bf t;
#pragma unroll
  for (int i = 0; i < 8; ++i) t[i] = (bf16)c[i];
  return __builtin_bit_cast(v4u, t);
}

// ---------------------------------------------------------------------------
// k5: one-time weight swizzle into B-fragment order, f32 -> bf16.
// dst[((kc*(N/16)+nt)*32+lane)*16+e] = src[(kc*32+(lane>>4)*16+e)*N + nt*16+(lane&15)]
// ---------------------------------------------------------------------------
static __device__ inline void swizzle(const float* __restrict__ src, bf16* __restrict__ dst,
                                      int K, int N, int tid, int nth) {
  int total = K * N;
  int ntiles = N >> 4;
  for (int idx = tid; idx < total; idx += nth) {
    int e     = idx & 15;
    int lane2 = (idx >> 4) & 31;
    int rest  = idx >> 9;
    int nt    = rest % ntiles;
    int kc    = rest / ntiles;
    int kk    = kc * 32 + (lane2 >> 4) * 16 + e;
    int col   = nt * 16 + (lane2 & 15);
    dst[idx]  = (bf16)src[kk * N + col];
  }
}

__global__ __launch_bounds__(256) void k5_prep(
    const float* __restrict__ Wk, const float* __restrict__ Wv,
    const float* __restrict__ Wq, const float* __restrict__ gK,
    const float* __restrict__ gR, const float* __restrict__ W1,
    const float* __restrict__ W2, bf16* __restrict__ Wkf, bf16* __restrict__ Wvf,
    bf16* __restrict__ Wqf, bf16* __restrict__ gKf, bf16* __restrict__ gRf,
    bf16* __restrict__ W1f, bf16* __restrict__ W2f) {
  int tid = blockIdx.x * blockDim.x + threadIdx.x;
  int nth = gridDim.x * blockDim.x;
  swizzle(Wk, Wkf, 64, 64, tid, nth);
  swizzle(Wv, Wvf, 64, 64, tid, nth);
  swizzle(Wq, Wqf, 64, 64, tid, nth);
  swizzle(gK, gKf, 64, 192, tid, nth);
  swizzle(gR, gRf, 64, 192, tid, nth);
  swizzle(W1, W1f, 64, 128, tid, nth);
  swizzle(W2, W2f, 128, 64, tid, nth);
}

// ---------------------------------------------------------------------------
// k0: slots = mu + exp(log_sigma) * noise   -> d_out
// ---------------------------------------------------------------------------
__global__ __launch_bounds__(256) void k0_init_slots(
    const float* __restrict__ noise, const float* __restrict__ mu,
    const float* __restrict__ logsig, float* __restrict__ slots) {
  int b = blockIdx.x;
  for (int i = threadIdx.x; i < NS * DSLOT; i += blockDim.x) {
    int d = i % DSLOT;
    slots[(size_t)b * NS * DSLOT + i] =
        mu[d] + __expf(logsig[d]) * noise[(size_t)b * NS * DSLOT + i];
  }
}

// ---------------------------------------------------------------------------
// k1: fused LayerNorm(inputs) + k/v projection. k token-major; v d-major.
// Weight B-frags come pre-swizzled from global (L2-resident).
// Block = 128 threads (4 waves); each wave does a 16x64 row tile.
// ---------------------------------------------------------------------------
__global__ __launch_bounds__(128) void k1_ln_project(
    const float* __restrict__ x, const float* __restrict__ gamma,
    const float* __restrict__ beta, const bf16* __restrict__ Wkf,
    const bf16* __restrict__ Wvf, bf16* __restrict__ kout, bf16* __restrict__ vT) {
  __shared__ bf16 lnbuf[4][16][DSLOT];  // 8 KB

  int wave = threadIdx.x >> 5, lane = threadIdx.x & 31;
  size_t row0 = (size_t)blockIdx.x * 64 + (size_t)wave * 16;
  size_t bidx = row0 / NTOK;
  int    tok0 = (int)(row0 % NTOK);

  int lr = lane >> 1, c0 = (lane & 1) * 32;
  const float* xp = x + (row0 + lr) * DSLOT + c0;
  float xv[32];
  float s0 = 0.f, s1 = 0.f;
#pragma unroll
  for (int i = 0; i < 32; ++i) { float t = xp[i]; xv[i] = t; s0 += t; s1 += t * t; }
  s0 += __shfl_xor(s0, 1, 32);
  s1 += __shfl_xor(s1, 1, 32);
  float mean = s0 * (1.f / 64.f);
  float var  = s1 * (1.f / 64.f) - mean * mean;
  float rstd = rsqrtf(var + 1e-3f);
#pragma unroll
  for (int i = 0; i < 32; ++i) {
    int c = c0 + i;
    lnbuf[wave][lr][c] = (bf16)((xv[i] - mean) * rstd * gamma[c] + beta[c]);
  }
  __syncthreads();

  v16bf a0 = load_a_frag_c(&lnbuf[wave][0][0], DSLOT, 0, lane);
  v16bf a1 = load_a_frag_c(&lnbuf[wave][0][0], DSLOT, 32, lane);
  int n = lane & 15, hf = lane >> 4;
  bf16* vTb = vT + bidx * (size_t)DSLOT * NTOK;
#pragma unroll
  for (int nt = 0; nt < 4; ++nt) {
    v8f ck = {};
    ck = wmma_bf16(a0, load_frag32(Wkf + ((0 * 4 + nt) * 32 + lane) * 16), ck);
    ck = wmma_bf16(a1, load_frag32(Wkf + ((1 * 4 + nt) * 32 + lane) * 16), ck);
    v8f cv = {};
    cv = wmma_bf16(a0, load_frag32(Wvf + ((0 * 4 + nt) * 32 + lane) * 16), cv);
    cv = wmma_bf16(a1, load_frag32(Wvf + ((1 * 4 + nt) * 32 + lane) * 16), cv);
#pragma unroll
    for (int i = 0; i < 8; ++i) {
      size_t r = row0 + i + 8 * hf;
      kout[r * DSLOT + nt * 16 + n] = (bf16)ck[i];
    }
    *(v4u*)(vTb + (size_t)(nt * 16 + n) * NTOK + tok0 + 8 * hf) = pack_bf16x8(cv);
  }
}

// ---------------------------------------------------------------------------
// k2: sn = LN(slots); q = (sn @ Wq) * scale, written in B-fragment order
// qf[b][kchunk][lane][16]. Also zeroes upd/csum. One wave per batch.
// ---------------------------------------------------------------------------
__global__ __launch_bounds__(32) void k2_q_project(
    const float* __restrict__ slots, const float* __restrict__ gamma,
    const float* __restrict__ beta, const bf16* __restrict__ Wqf,
    bf16* __restrict__ qf, float* __restrict__ upd, float* __restrict__ csum) {
  __shared__ bf16 sn[SPAD][DSLOT];
  int b = blockIdx.x, lane = threadIdx.x;

  for (int i = lane; i < SPAD * DSLOT; i += 32) upd[(size_t)b * SPAD * DSLOT + i] = 0.f;
  if (lane < SPAD) csum[b * SPAD + lane] = 0.f;

  int lr = lane >> 1, c0 = (lane & 1) * 32;
  float xv[32];
  float s0 = 0.f, s1 = 0.f;
  if (lr < NS) {
    const float* sp = slots + ((size_t)b * NS + lr) * DSLOT + c0;
#pragma unroll
    for (int i = 0; i < 32; ++i) { float t = sp[i]; xv[i] = t; s0 += t; s1 += t * t; }
  }
  s0 += __shfl_xor(s0, 1, 32);
  s1 += __shfl_xor(s1, 1, 32);
  if (lr < NS) {
    float mean = s0 * (1.f / 64.f);
    float var  = s1 * (1.f / 64.f) - mean * mean;
    float rstd = rsqrtf(var + 1e-3f);
#pragma unroll
    for (int i = 0; i < 32; ++i) {
      int c = c0 + i;
      sn[lr][c] = (bf16)((xv[i] - mean) * rstd * gamma[c] + beta[c]);
    }
  } else {
#pragma unroll
    for (int i = 0; i < 32; ++i) sn[lr][c0 + i] = (bf16)0.f;
  }
  __syncthreads();

  v16bf a0 = load_a_frag_c(&sn[0][0], DSLOT, 0, lane);
  v16bf a1 = load_a_frag_c(&sn[0][0], DSLOT, 32, lane);
  int n = lane & 15, hf = lane >> 4;
#pragma unroll
  for (int nt = 0; nt < 4; ++nt) {
    v8f c = {};
    c = wmma_bf16(a0, load_frag32(Wqf + ((0 * 4 + nt) * 32 + lane) * 16), c);
    c = wmma_bf16(a1, load_frag32(Wqf + ((1 * 4 + nt) * 32 + lane) * 16), c);
#pragma unroll
    for (int i = 0; i < 8; ++i) {
      int dcol = nt * 16 + n;   // K index (d)
      int srow = i + 8 * hf;    // N index (slot)
      int kc   = dcol >> 5;
      int rem  = dcol & 31;
      qf[((size_t)b * 2 + kc) * 512 + ((rem >> 4) * 16 + srow) * 16 + (rem & 15)] =
          (bf16)(c[i] * 0.125f);
    }
  }
}

// ---------------------------------------------------------------------------
// k3: logits -> masked softmax over 11 slots -> (attn+eps)^T @ v partials.
// Renorm folded (applied in k4). Grid = NBATCH*NSPLIT, 128 threads (4 waves).
// ---------------------------------------------------------------------------
__global__ __launch_bounds__(128) void k3_attention(
    const bf16* __restrict__ kmat, const bf16* __restrict__ vT,
    const bf16* __restrict__ qf, float* __restrict__ upd, float* __restrict__ csum) {
  __shared__ bf16 attA[4][32][16];  // attn^T A-fragments, per wave

  int b = blockIdx.x / NSPLIT, split = blockIdx.x % NSPLIT;
  int wave = threadIdx.x >> 5, lane = threadIdx.x & 31;
  int n = lane & 15, hf = lane >> 4;

  const bf16* kb  = kmat + (size_t)b * NTOK * DSLOT;
  const bf16* vTb = vT + (size_t)b * DSLOT * NTOK;

  v16bf bq0 = load_frag32(qf + ((size_t)b * 2 + 0) * 512 + lane * 16);
  v16bf bq1 = load_frag32(qf + ((size_t)b * 2 + 1) * 512 + lane * 16);

  v8f U[4] = {};
  float colacc = 0.f;
  bool valid = (n < NS);

  const int rows_per_block = NTOK / NSPLIT;  // 512
  for (int tile = 0; tile < rows_per_block / 128; ++tile) {
    int n0 = split * rows_per_block + tile * 128 + wave * 32;
    __builtin_prefetch(kb + (size_t)(n0 + 128) * DSLOT, 0, 1);
    __builtin_prefetch(vTb + n0 + 128, 0, 1);

#pragma unroll
    for (int sub = 0; sub < 2; ++sub) {
      int nr = n0 + sub * 16;
      v16bf a0 = load_a_frag_c(kb + (size_t)nr * DSLOT, DSLOT, 0, lane);
      v16bf a1 = load_a_frag_c(kb + (size_t)nr * DSLOT, DSLOT, 32, lane);
      v8f lg = {};
      lg = wmma_bf16(a0, bq0, lg);
      lg = wmma_bf16(a1, bq1, lg);
#pragma unroll
      for (int i = 0; i < 8; ++i) {
        float vmax = valid ? lg[i] : -1e30f;
#pragma unroll
        for (int m = 1; m < 16; m <<= 1) vmax = fmaxf(vmax, __shfl_xor(vmax, m, 32));
        float e = valid ? __expf(lg[i] - vmax) : 0.f;
        float sm = e;
#pragma unroll
        for (int m = 1; m < 16; m <<= 1) sm += __shfl_xor(sm, m, 32);
        float a = valid ? (e / sm + 1e-8f) : 0.f;
        colacc += a;
        int nn = sub * 16 + i + 8 * hf;
        attA[wave][(nn >> 4) * 16 + n][nn & 15] = (bf16)a;
      }
    }
    __syncthreads();
    v16bf at = load_frag32(&attA[wave][lane][0]);
#pragma unroll
    for (int dt = 0; dt < 4; ++dt) {
      v16bf bv = load_frag32(vTb + (size_t)(dt * 16 + n) * NTOK + n0 + hf * 16);
      U[dt] = wmma_bf16(at, bv, U[dt]);
    }
    __syncthreads();
  }

  float* updb = upd + (size_t)b * SPAD * DSLOT;
#pragma unroll
  for (int dt = 0; dt < 4; ++dt)
#pragma unroll
    for (int i = 0; i < 8; ++i)
      atomicAdd(&updb[(i + 8 * hf) * DSLOT + dt * 16 + n], U[dt][i]);

  colacc += __shfl_xor(colacc, 16, 32);
  if (hf == 0 && valid) atomicAdd(&csum[b * SPAD + n], colacc);
}

// ---------------------------------------------------------------------------
// k4: finalize renorm, GRU (reset_after), LayerNorm, MLP, residual — now all
// three GEMMs on the WMMA pipe. One wave per batch; A-operands staged bf16 in
// LDS, B-operands are pre-swizzled weight fragments from global.
// ---------------------------------------------------------------------------
__global__ __launch_bounds__(32) void k4_gru_mlp(
    const float* __restrict__ upd_raw, const float* __restrict__ csum,
    const bf16* __restrict__ gKf, const bf16* __restrict__ gRf,
    const float* __restrict__ gbias, const float* __restrict__ g_mlp,
    const float* __restrict__ b_mlp, const bf16* __restrict__ W1f,
    const float* __restrict__ b1, const bf16* __restrict__ W2f,
    const float* __restrict__ b2, float* __restrict__ slots) {
  __shared__ bf16  uA[SPAD][DSLOT];        // updates (A operand)
  __shared__ bf16  hA[SPAD][DSLOT];        // prev slots (A operand)
  __shared__ float hp32[NS][DSLOT];
  __shared__ float mx[SPAD][3 * DSLOT];
  __shared__ float mh[SPAD][3 * DSLOT];
  __shared__ float hn32[NS][DSLOT];
  __shared__ bf16  lnA[SPAD][DSLOT];
  __shared__ bf16  hidA[SPAD][HID];
  __shared__ float rm[NS], rs[NS];

  int b = blockIdx.x, lane = threadIdx.x;
  int n = lane & 15, hf = lane >> 4;

  for (int i = lane; i < SPAD * DSLOT; i += 32) {
    int s = i >> 6, c = i & 63;
    float uv = 0.f, hv = 0.f;
    if (s < NS) {
      uv = upd_raw[(size_t)b * SPAD * DSLOT + i] / csum[b * SPAD + s];
      hv = slots[(size_t)b * NS * DSLOT + s * DSLOT + c];
      hp32[s][c] = hv;
    }
    uA[s][c] = (bf16)uv;
    hA[s][c] = (bf16)hv;
  }
  __syncthreads();

  // GRU input projections: mx = u @ gK + bi ; mh = h @ gR + br
  v16bf au0 = load_a_frag_c(&uA[0][0], DSLOT, 0, lane);
  v16bf au1 = load_a_frag_c(&uA[0][0], DSLOT, 32, lane);
  v16bf ahp0 = load_a_frag_c(&hA[0][0], DSLOT, 0, lane);
  v16bf ahp1 = load_a_frag_c(&hA[0][0], DSLOT, 32, lane);
#pragma unroll
  for (int nt = 0; nt < 12; ++nt) {
    v8f cx = {};
    cx = wmma_bf16(au0, load_frag32(gKf + ((0 * 12 + nt) * 32 + lane) * 16), cx);
    cx = wmma_bf16(au1, load_frag32(gKf + ((1 * 12 + nt) * 32 + lane) * 16), cx);
    v8f ch = {};
    ch = wmma_bf16(ahp0, load_frag32(gRf + ((0 * 12 + nt) * 32 + lane) * 16), ch);
    ch = wmma_bf16(ahp1, load_frag32(gRf + ((1 * 12 + nt) * 32 + lane) * 16), ch);
    int c = nt * 16 + n;
#pragma unroll
    for (int i = 0; i < 8; ++i) {
      mx[i + 8 * hf][c] = cx[i] + gbias[c];
      mh[i + 8 * hf][c] = ch[i] + gbias[3 * DSLOT + c];
    }
  }
  __syncthreads();

  // Gates
  for (int i = lane; i < NS * DSLOT; i += 32) {
    int s = i >> 6, c = i & 63;
    float z = 1.f / (1.f + __expf(-(mx[s][c] + mh[s][c])));
    float r = 1.f / (1.f + __expf(-(mx[s][DSLOT + c] + mh[s][DSLOT + c])));
    float cand = tanhf(mx[s][2 * DSLOT + c] + r * mh[s][2 * DSLOT + c]);
    hn32[s][c] = z * hp32[s][c] + (1.f - z) * cand;
  }
  __syncthreads();

  if (lane < NS) {
    float s0 = 0.f, s1 = 0.f;
    for (int d = 0; d < DSLOT; ++d) { float xx = hn32[lane][d]; s0 += xx; s1 += xx * xx; }
    float mean = s0 * (1.f / 64.f), var = s1 * (1.f / 64.f) - mean * mean;
    rm[lane] = mean;
    rs[lane] = rsqrtf(var + 1e-3f);
  }
  __syncthreads();

  for (int i = lane; i < SPAD * DSLOT; i += 32) {
    int s = i >> 6, c = i & 63;
    lnA[s][c] = (s < NS)
        ? (bf16)((hn32[s][c] - rm[s]) * rs[s] * g_mlp[c] + b_mlp[c])
        : (bf16)0.f;
  }
  __syncthreads();

  // MLP layer 1: hid = relu(ln @ W1 + b1)
  v16bf al0 = load_a_frag_c(&lnA[0][0], DSLOT, 0, lane);
  v16bf al1 = load_a_frag_c(&lnA[0][0], DSLOT, 32, lane);
#pragma unroll
  for (int nt = 0; nt < 8; ++nt) {
    v8f c1 = {};
    c1 = wmma_bf16(al0, load_frag32(W1f + ((0 * 8 + nt) * 32 + lane) * 16), c1);
    c1 = wmma_bf16(al1, load_frag32(W1f + ((1 * 8 + nt) * 32 + lane) * 16), c1);
    int j = nt * 16 + n;
#pragma unroll
    for (int i = 0; i < 8; ++i)
      hidA[i + 8 * hf][j] = (bf16)fmaxf(c1[i] + b1[j], 0.f);
  }
  __syncthreads();

  // MLP layer 2 + residual: slots = hn + hid @ W2 + b2
  v16bf a2[4];
#pragma unroll
  for (int kc = 0; kc < 4; ++kc) a2[kc] = load_a_frag_c(&hidA[0][0], HID, kc * 32, lane);
#pragma unroll
  for (int nt = 0; nt < 4; ++nt) {
    v8f c2 = {};
#pragma unroll
    for (int kc = 0; kc < 4; ++kc)
      c2 = wmma_bf16(a2[kc], load_frag32(W2f + ((kc * 4 + nt) * 32 + lane) * 16), c2);
    int c = nt * 16 + n;
#pragma unroll
    for (int i = 0; i < 8; ++i) {
      int s = i + 8 * hf;
      if (s < NS)
        slots[(size_t)b * NS * DSLOT + s * DSLOT + c] = hn32[s][c] + c2[i] + b2[c];
    }
  }
}

// ---------------------------------------------------------------------------
extern "C" void kernel_launch(void* const* d_in, const int* in_sizes, int n_in,
                              void* d_out, int out_size, void* d_ws, size_t ws_size,
                              hipStream_t stream) {
  const float* inputs       = (const float*)d_in[0];
  const float* noise        = (const float*)d_in[1];
  const float* gamma_in     = (const float*)d_in[2];
  const float* beta_in      = (const float*)d_in[3];
  const float* gamma_slots  = (const float*)d_in[4];
  const float* beta_slots   = (const float*)d_in[5];
  const float* gamma_mlp    = (const float*)d_in[6];
  const float* beta_mlp     = (const float*)d_in[7];
  const float* slots_mu     = (const float*)d_in[8];
  const float* slots_ls     = (const float*)d_in[9];
  const float* Wq           = (const float*)d_in[10];
  const float* Wk           = (const float*)d_in[11];
  const float* Wv           = (const float*)d_in[12];
  const float* gru_kernel   = (const float*)d_in[13];
  const float* gru_rkernel  = (const float*)d_in[14];
  const float* gru_bias     = (const float*)d_in[15];
  const float* mlp_W1       = (const float*)d_in[16];
  const float* mlp_b1       = (const float*)d_in[17];
  const float* mlp_W2       = (const float*)d_in[18];
  const float* mlp_b2       = (const float*)d_in[19];
  float* slots = (float*)d_out;

  bf16* kbuf = (bf16*)d_ws;
  bf16* vT   = kbuf + (size_t)NBATCH * NTOK * DSLOT;
  bf16* qf   = vT + (size_t)NBATCH * NTOK * DSLOT;
  bf16* Wkf  = qf + (size_t)NBATCH * 2 * 512;
  bf16* Wvf  = Wkf + 4096;
  bf16* Wqf  = Wvf + 4096;
  bf16* gKf  = Wqf + 4096;
  bf16* gRf  = gKf + 12288;
  bf16* W1f  = gRf + 12288;
  bf16* W2f  = W1f + 8192;
  float* upd = (float*)(W2f + 8192);
  float* cs  = upd + (size_t)NBATCH * SPAD * DSLOT;

  k5_prep<<<64, 256, 0, stream>>>(Wk, Wv, Wq, gru_kernel, gru_rkernel, mlp_W1,
                                  mlp_W2, Wkf, Wvf, Wqf, gKf, gRf, W1f, W2f);
  k0_init_slots<<<NBATCH, 256, 0, stream>>>(noise, slots_mu, slots_ls, slots);
  k1_ln_project<<<(NBATCH * NTOK) / 64, 128, 0, stream>>>(inputs, gamma_in, beta_in,
                                                          Wkf, Wvf, kbuf, vT);
  for (int it = 0; it < 3; ++it) {
    k2_q_project<<<NBATCH, 32, 0, stream>>>(slots, gamma_slots, beta_slots, Wqf, qf, upd, cs);
    k3_attention<<<NBATCH * NSPLIT, 128, 0, stream>>>(kbuf, vT, qf, upd, cs);
    k4_gru_mlp<<<NBATCH, 32, 0, stream>>>(upd, cs, gKf, gRf, gru_bias,
                                          gamma_mlp, beta_mlp, W1f, mlp_b1,
                                          W2f, mlp_b2, slots);
  }
}
